// SwinTransformerBlock_27891517620498
// MI455X (gfx1250) — compile-verified, matrix operand order checked by reading, not run on hardware
//
#include <hip/hip_runtime.h>
#include <hip/hip_bf16.h>

typedef __bf16 bf16_t;
typedef __attribute__((ext_vector_type(16))) __bf16 v16bf;
typedef __attribute__((ext_vector_type(8)))  float  v8f;
typedef unsigned int v4u  __attribute__((ext_vector_type(4)));
typedef int          v8i_ __attribute__((ext_vector_type(8)));
typedef int          v4i_ __attribute__((ext_vector_type(4)));

#define WSZ     7
#define SHIFT3  3
#define SDIM    28
#define HDIM    28
#define WDIM    56
#define CDIM    192
#define NHEAD   6
#define HD      32
#define NTOK    343
#define NPAD    352
#define HID     768
#define NWIN_B  128
#define BWIN    256
#define TOK     43904      /* S*H*W */
#define ROWS    87808      /* 2*TOK */
#define SCALE_Q 0.17677669529663687f
#define EPS_LN  1e-5f

#if defined(__has_builtin)
#if __has_builtin(__builtin_amdgcn_tensor_load_to_lds) && __has_builtin(__builtin_amdgcn_s_wait_tensorcnt)
#define HAVE_TDM 1
#endif
#endif

union Frag { v16bf v; uint4 q[2]; };

__device__ inline v16bf load_frag(const bf16_t* base, int ld, int kk) {
    int lane = threadIdx.x & 31;
    const bf16_t* p = base + (long)(lane & 15) * ld + kk + ((lane < 16) ? 0 : 8);
    Frag f;
    f.q[0] = *(const uint4*)(p);
    f.q[1] = *(const uint4*)(p + 16);
    return f.v;
}

__device__ inline v8f wmma_bf16(v16bf a, v16bf b, v8f c) {
    return __builtin_amdgcn_wmma_f32_16x16x32_bf16(false, a, false, b, (short)0, c, false, false);
}

__device__ inline float wave_sum(float v) {
    for (int o = 16; o > 0; o >>= 1) v += __shfl_xor(v, o, 32);
    return v;
}
__device__ inline float wave_max(float v) {
    for (int o = 16; o > 0; o >>= 1) v = fmaxf(v, __shfl_xor(v, o, 32));
    return v;
}

#ifdef HAVE_TDM
/* 1-D contiguous global->LDS tile copy through the Tensor Data Mover.
 * D# per CDNA5 ISA ch.8: group0 = {count/flags, lds_addr, global_addr, type=2},
 * group1 = {data_size=2B, tensor_dim0, tensor_dim1=1, tile_dim0, stride}.
 * This toolchain exposes the 6-arg builtin: (g0, g1, g2, g3, g4, cpol).
 * Issue from ONE wave only; completion tracked with TENSORcnt. */
__device__ inline void tdm_load_1d_bf16(unsigned int lds_addr, const void* gptr,
                                        unsigned int nelem) {
    unsigned long long ga = (unsigned long long)(size_t)gptr;
    v4u g0;
    g0[0] = 1u;                                              /* count=1 */
    g0[1] = lds_addr;                                        /* lds_addr  [63:32]  */
    g0[2] = (unsigned int)ga;                                /* gaddr lo  [95:64]  */
    g0[3] = (unsigned int)((ga >> 32) & 0x01FFFFFFu) | 0x80000000u; /* gaddr hi + type=2 */
    v8i_ g1;
    g1[0] = 0x10000;                                         /* data_size=2 bytes  */
    g1[1] = (int)((nelem & 0xFFFFu) << 16);                  /* tensor_dim0 lo16   */
    g1[2] = (int)((nelem >> 16) | (1u << 16));               /* tensor_dim0 hi16, tensor_dim1=1 */
    g1[3] = (int)((nelem & 0xFFFFu) << 16);                  /* tile_dim0 [127:112] */
    g1[4] = 0;                                               /* tile_dim1=0, tile_dim2=0 */
    g1[5] = (int)nelem;                                      /* tensor_dim0_stride lo32 */
    g1[6] = 0;
    g1[7] = 0;
    v4i_ z4 = {0, 0, 0, 0};
    v8i_ z8 = {0, 0, 0, 0, 0, 0, 0, 0};
    __builtin_amdgcn_tensor_load_to_lds(g0, g1, z4, z4, z8, 0);
}
#endif

/* ---------------- weight convert: dst[c][r] = bf16(src[r][c]) ---------------- */
__global__ void k_wt(const float* __restrict__ src, bf16_t* __restrict__ dst, int R, int Cc) {
    int i = blockIdx.x * 256 + threadIdx.x;
    if (i >= R * Cc) return;
    int r = i / Cc, c = i % Cc;
    dst[(long)c * R + r] = (bf16_t)src[i];
}

/* -------- fused LN + roll + window-gather + QKV GEMM (one block per window) -------- */
__global__ __launch_bounds__(256) void k_qkv(
    const float* __restrict__ x1, const float* __restrict__ x2,
    const float* __restrict__ g1, const float* __restrict__ b1,
    const float* __restrict__ g2, const float* __restrict__ b2,
    const bf16_t* __restrict__ wqT, const float* __restrict__ bq,
    const bf16_t* __restrict__ wkvT, const float* __restrict__ bkv,
    bf16_t* __restrict__ qg, bf16_t* __restrict__ kg, bf16_t* __restrict__ vtg)
{
    extern __shared__ char smem[];
    bf16_t* aw = (bf16_t*)smem;          // [NPAD][CDIM]
    bf16_t* cw = aw + NPAD * CDIM;       // [NPAD][CDIM]

    int w  = blockIdx.x;
    int b  = w >> 7, wl = w & 127;
    int wz = wl >> 5, wy = (wl >> 3) & 3, wx = wl & 7;
    int lane = threadIdx.x & 31, wave = threadIdx.x >> 5;

    for (int t = wave; t < NPAD; t += 8) {
        if (t >= NTOK) {
            for (int kq = 0; kq < 6; ++kq) {
                aw[(long)t * CDIM + lane + 32 * kq] = (bf16_t)0.f;
                cw[(long)t * CDIM + lane + 32 * kq] = (bf16_t)0.f;
            }
            continue;
        }
        int z = t / 49, y = (t / 7) % 7, xx = t % 7;
        int sz = (wz * 7 + z + SHIFT3) % SDIM;
        int sy = (wy * 7 + y + SHIFT3) % HDIM;
        int sx = (wx * 7 + xx + SHIFT3) % WDIM;
        long src = ((long)b * TOK + (long)sz * (HDIM * WDIM) + sy * WDIM + sx) * CDIM;

        float v[6], u[6], m1 = 0.f, m2 = 0.f;
        for (int kq = 0; kq < 6; ++kq) {
            v[kq] = x1[src + lane + 32 * kq]; m1 += v[kq];
            u[kq] = x2[src + lane + 32 * kq]; m2 += u[kq];
        }
        m1 = wave_sum(m1) * (1.f / CDIM);
        m2 = wave_sum(m2) * (1.f / CDIM);
        float s1 = 0.f, s2 = 0.f;
        for (int kq = 0; kq < 6; ++kq) {
            float d1 = v[kq] - m1, d2 = u[kq] - m2;
            s1 += d1 * d1; s2 += d2 * d2;
        }
        float rs1 = rsqrtf(wave_sum(s1) * (1.f / CDIM) + EPS_LN);
        float rs2 = rsqrtf(wave_sum(s2) * (1.f / CDIM) + EPS_LN);
        for (int kq = 0; kq < 6; ++kq) {
            int c = lane + 32 * kq;
            aw[(long)t * CDIM + c] = (bf16_t)((v[kq] - m1) * rs1 * g1[c] + b1[c]);
            cw[(long)t * CDIM + c] = (bf16_t)((u[kq] - m2) * rs2 * g2[c] + b2[c]);
        }
    }
    __syncthreads();

    int Nc = lane & 15, rowoff = (lane < 16) ? 0 : 8;
    for (int tile = wave; tile < 22 * 36; tile += 8) {
        int rt = tile / 36, ct = tile % 36;
        const bf16_t* A; const bf16_t* Brow;
        if (ct < 12)      { A = aw; Brow = wqT  + (long)(ct * 16) * CDIM; }
        else if (ct < 24) { A = cw; Brow = wkvT + (long)((ct - 12) * 16) * CDIM; }
        else              { A = cw; Brow = wkvT + (long)(192 + (ct - 24) * 16) * CDIM; }

        v8f acc = {};
        for (int ks = 0; ks < 6; ++ks) {
            v16bf af  = load_frag(A + (long)rt * 16 * CDIM, CDIM, ks * 32);
            v16bf bfg = load_frag(Brow, CDIM, ks * 32);
            acc = wmma_bf16(af, bfg, acc);
        }
        if (ct < 12) {
            int j = ct * 16 + Nc, head = j >> 5, d = j & 31;
            float bb = bq[j];
            bf16_t* dst = qg + ((long)w * NHEAD + head) * (NPAD * HD) + d;
            for (int e = 0; e < 8; ++e) {
                int n = rt * 16 + e + rowoff;
                dst[(long)n * HD] = (bf16_t)((acc[e] + bb) * SCALE_Q);
            }
        } else if (ct < 24) {
            int j = (ct - 12) * 16 + Nc, head = j >> 5, d = j & 31;
            float bb = bkv[j];
            bf16_t* dst = kg + ((long)w * NHEAD + head) * (NPAD * HD) + d;
            for (int e = 0; e < 8; ++e) {
                int n = rt * 16 + e + rowoff;
                dst[(long)n * HD] = (bf16_t)(acc[e] + bb);
            }
        } else {
            int j = (ct - 24) * 16 + Nc, head = j >> 5, d = j & 31;
            float bb = bkv[192 + j];
            int n0 = rt * 16 + rowoff;
            alignas(16) bf16_t tmp[8];
            for (int e = 0; e < 8; ++e) tmp[e] = (bf16_t)(acc[e] + bb);
            bf16_t* dst = vtg + (((long)w * NHEAD + head) * HD + d) * NPAD + n0;
            *(uint4*)dst = *(const uint4*)tmp;
        }
    }
}

/* ---------------- attention: one block per (window, head), 4 waves ---------------- */
__global__ __launch_bounds__(128) void k_attn(
    const bf16_t* __restrict__ qg, const bf16_t* __restrict__ kg,
    const bf16_t* __restrict__ vtg,
    const float* __restrict__ bias_table, const float* __restrict__ mask,
    bf16_t* __restrict__ owin)
{
    extern __shared__ char smem[];
    bf16_t* ksh = (bf16_t*)smem;                       // [NPAD][HD]
    bf16_t* vsh = ksh + NPAD * HD;                     // [HD][NPAD]
    float*  scores = (float*)(vsh + NPAD * HD);        // 4 * [16][NPAD]
    bf16_t* probs  = (bf16_t*)(scores + 4 * 16 * NPAD);// 4 * [16][NPAD]

    int wh = blockIdx.x;
    int w = wh / NHEAD, h = wh % NHEAD;
    int wl = w & 127;
    int tid = threadIdx.x, lane = tid & 31, wave = tid >> 5;

    long base = ((long)w * NHEAD + h) * (NPAD * HD);
#ifdef HAVE_TDM
    if (wave == 0) {
        tdm_load_1d_bf16((unsigned int)(size_t)(void*)ksh, kg + base, NPAD * HD);
        tdm_load_1d_bf16((unsigned int)(size_t)(void*)vsh, vtg + base, NPAD * HD);
        __builtin_amdgcn_s_wait_tensorcnt(0);
    }
#else
    {
        const uint4* ks4 = (const uint4*)(kg + base);
        const uint4* vs4 = (const uint4*)(vtg + base);
        uint4* dk = (uint4*)ksh; uint4* dv = (uint4*)vsh;
        for (int i = tid; i < NPAD * HD / 8; i += 128) { dk[i] = ks4[i]; dv[i] = vs4[i]; }
    }
#endif
    __syncthreads();

    float*  my_sc = scores + (long)wave * 16 * NPAD;
    bf16_t* my_pr = probs  + (long)wave * 16 * NPAD;
    const bf16_t* qbase = qg + base;
    int Nc = lane & 15, rowoff = (lane < 16) ? 0 : 8;

    for (int qt = wave; qt < 22; qt += 4) {
        v16bf af = load_frag(qbase + (long)qt * 16 * HD, HD, 0);

        /* per-row precompute: n coordinates -> packed rel-pos row code */
        int rowcode[8]; bool rowok[8];
        for (int e = 0; e < 8; ++e) {
            int n = qt * 16 + e + rowoff;
            rowok[e] = (n < NTOK);
            int nn = rowok[e] ? n : 0;
            rowcode[e] = (nn / 49) * 20 + ((nn / 7) % 7) * 13 + (nn % 7) + 204;
        }

        for (int kt = 0; kt < 22; ++kt) {
            v16bf bfg = load_frag(ksh + (long)kt * 16 * HD, HD, 0);
            v8f acc = {};
            acc = wmma_bf16(af, bfg, acc);
            int m = kt * 16 + Nc;
            bool mok = (m < NTOK);
            int mm = mok ? m : 0;
            int colcode = (mm / 49) * 20 + ((mm / 7) % 7) * 13 + (mm % 7);
            const float* mcol = mask + (long)wl * NTOK * NTOK + m;
            for (int e = 0; e < 8; ++e) {
                int n = qt * 16 + e + rowoff;
                float sval;
                if (rowok[e] && mok) {
                    int idx = rowcode[e] - colcode;
                    sval = acc[e] + bias_table[(long)idx * NHEAD + h]
                                  + mcol[(long)n * NTOK];
                } else {
                    sval = -1e30f;
                }
                my_sc[(long)(e + rowoff) * NPAD + m] = sval;
            }
        }
        /* row softmax (per wave, rows 0..15 of this q-tile) */
        for (int r = 0; r < 16; ++r) {
            float* row = my_sc + (long)r * NPAD;
            float mx = -1e30f;
            for (int i = lane; i < NPAD; i += 32) mx = fmaxf(mx, row[i]);
            mx = wave_max(mx);
            float s = 0.f;
            for (int i = lane; i < NPAD; i += 32) {
                float e = __expf(row[i] - mx);
                row[i] = e; s += e;
            }
            s = wave_sum(s);
            float inv = 1.f / s;
            bf16_t* prow = my_pr + (long)r * NPAD;
            for (int i = lane; i < NPAD; i += 32) prow[i] = (bf16_t)(row[i] * inv);
        }
        /* PV: out[16 x 32] */
        for (int dt = 0; dt < 2; ++dt) {
            v8f acc = {};
            for (int it = 0; it < 11; ++it) {
                v16bf pf = load_frag(my_pr, NPAD, it * 32);
                v16bf vf = load_frag(vsh + (long)(dt * 16) * NPAD, NPAD, it * 32);
                acc = wmma_bf16(pf, vf, acc);
            }
            int j = h * HD + dt * 16 + Nc;
            bf16_t* dst = owin + (long)w * NPAD * CDIM + j;
            for (int e = 0; e < 8; ++e) {
                int n = qt * 16 + e + rowoff;
                dst[(long)n * CDIM] = (bf16_t)acc[e];
            }
        }
    }
}

/* ---------- projection GEMM + inverse roll scatter + residual (block/window) ---------- */
__global__ __launch_bounds__(256) void k_proj(
    const bf16_t* __restrict__ owin, const bf16_t* __restrict__ wpT,
    const float* __restrict__ bp, const float* __restrict__ x1,
    float* __restrict__ xout)
{
    extern __shared__ char smem[];
    bf16_t* osh = (bf16_t*)smem;  // [NPAD][CDIM]
    int w = blockIdx.x, b = w >> 7, wl = w & 127;
    int wz = wl >> 5, wy = (wl >> 3) & 3, wx = wl & 7;
    int tid = threadIdx.x, lane = tid & 31, wave = tid >> 5;

    const uint4* s4 = (const uint4*)(owin + (long)w * NPAD * CDIM);
    uint4* d4 = (uint4*)osh;
    for (int i = tid; i < NPAD * CDIM / 8; i += 256) d4[i] = s4[i];
    __syncthreads();

    int Nc = lane & 15, rowoff = (lane < 16) ? 0 : 8;
    for (int tile = wave; tile < 22 * 12; tile += 8) {
        int rt = tile / 12, ct = tile % 12;
        v8f acc = {};
        for (int ks = 0; ks < 6; ++ks) {
            v16bf af  = load_frag(osh + (long)rt * 16 * CDIM, CDIM, ks * 32);
            v16bf bfg = load_frag(wpT + (long)(ct * 16) * CDIM, CDIM, ks * 32);
            acc = wmma_bf16(af, bfg, acc);
        }
        int j = ct * 16 + Nc;
        float bb = bp[j];
        for (int e = 0; e < 8; ++e) {
            int n = rt * 16 + e + rowoff;
            if (n >= NTOK) continue;
            int z = n / 49, y = (n / 7) % 7, xx = n % 7;
            int ds = (wz * 7 + z + SHIFT3) % SDIM;
            int dh = (wy * 7 + y + SHIFT3) % HDIM;
            int dw = (wx * 7 + xx + SHIFT3) % WDIM;
            long addr = ((long)b * TOK + (long)ds * (HDIM * WDIM) + dh * WDIM + dw) * CDIM + j;
            xout[addr] = x1[addr] + acc[e] + bb;
        }
    }
}

/* ---------------- LN2 -> bf16 ---------------- */
__global__ __launch_bounds__(256) void k_ln2(
    const float* __restrict__ x, const float* __restrict__ g,
    const float* __restrict__ bb, bf16_t* __restrict__ xn2)
{
    long row = (long)blockIdx.x * 8 + (threadIdx.x >> 5);
    int lane = threadIdx.x & 31;
    const float* src = x + row * CDIM;
    float v[6], m = 0.f;
    for (int kq = 0; kq < 6; ++kq) { v[kq] = src[lane + 32 * kq]; m += v[kq]; }
    m = wave_sum(m) * (1.f / CDIM);
    float s = 0.f;
    for (int kq = 0; kq < 6; ++kq) { float d = v[kq] - m; s += d * d; }
    float rs = rsqrtf(wave_sum(s) * (1.f / CDIM) + EPS_LN);
    for (int kq = 0; kq < 6; ++kq) {
        int c = lane + 32 * kq;
        xn2[row * CDIM + c] = (bf16_t)((v[kq] - m) * rs * g[c] + bb[c]);
    }
}

/* -------- fused MLP (64 rows / block): GELU hidden in LDS, final residual -------- */
__global__ __launch_bounds__(256) void k_mlp(
    const bf16_t* __restrict__ xn2, const bf16_t* __restrict__ wf1T,
    const float* __restrict__ bf1, const bf16_t* __restrict__ wf2T,
    const float* __restrict__ bf2, const float* __restrict__ x,
    float* __restrict__ out)
{
    extern __shared__ char smem[];
    bf16_t* xsh = (bf16_t*)smem;       // [64][CDIM]
    bf16_t* hsh = xsh + 64 * CDIM;     // [64][HID]
    long row0 = (long)blockIdx.x * 64;
    int tid = threadIdx.x, lane = tid & 31, wave = tid >> 5;

    const uint4* s4 = (const uint4*)(xn2 + row0 * CDIM);
    uint4* d4 = (uint4*)xsh;
    for (int i = tid; i < 64 * CDIM / 8; i += 256) d4[i] = s4[i];
    __syncthreads();

    int Nc = lane & 15, rowoff = (lane < 16) ? 0 : 8;
    for (int tile = wave; tile < 4 * 48; tile += 8) {
        int rt = tile / 48, ct = tile % 48;
        v8f acc = {};
        for (int ks = 0; ks < 6; ++ks) {
            v16bf af  = load_frag(xsh + (long)rt * 16 * CDIM, CDIM, ks * 32);
            v16bf bfg = load_frag(wf1T + (long)(ct * 16) * CDIM, CDIM, ks * 32);
            acc = wmma_bf16(af, bfg, acc);
        }
        int j = ct * 16 + Nc;
        float bb = bf1[j];
        for (int e = 0; e < 8; ++e) {
            float vv = acc[e] + bb;
            vv = 0.5f * vv * (1.f + erff(vv * 0.7071067811865476f));
            hsh[(long)(rt * 16 + e + rowoff) * HID + j] = (bf16_t)vv;
        }
    }
    __syncthreads();

    for (int tile = wave; tile < 4 * 12; tile += 8) {
        int rt = tile / 12, ct = tile % 12;
        v8f acc = {};
        for (int ks = 0; ks < 24; ++ks) {
            v16bf af  = load_frag(hsh + (long)rt * 16 * HID, HID, ks * 32);
            v16bf bfg = load_frag(wf2T + (long)(ct * 16) * HID, HID, ks * 32);
            acc = wmma_bf16(af, bfg, acc);
        }
        int j = ct * 16 + Nc;
        float bb = bf2[j];
        for (int e = 0; e < 8; ++e) {
            long r = row0 + rt * 16 + e + rowoff;
            out[r * CDIM + j] = x[r * CDIM + j] + acc[e] + bb;
        }
    }
}

extern "C" void kernel_launch(void* const* d_in, const int* in_sizes, int n_in,
                              void* d_out, int out_size, void* d_ws, size_t ws_size,
                              hipStream_t stream) {
    const float* x1   = (const float*)d_in[0];
    const float* x2   = (const float*)d_in[1];
    const float* mask = (const float*)d_in[2];
    const float* g1   = (const float*)d_in[3];
    const float* b1   = (const float*)d_in[4];
    const float* g2   = (const float*)d_in[5];
    const float* b2   = (const float*)d_in[6];
    const float* wq   = (const float*)d_in[7];
    const float* bq   = (const float*)d_in[8];
    const float* wkv  = (const float*)d_in[9];
    const float* bkv  = (const float*)d_in[10];
    const float* bias_table = (const float*)d_in[11];
    const float* wp   = (const float*)d_in[12];
    const float* bp   = (const float*)d_in[13];
    const float* gn2  = (const float*)d_in[14];
    const float* bn2  = (const float*)d_in[15];
    const float* wf1  = (const float*)d_in[16];
    const float* bf1  = (const float*)d_in[17];
    const float* wf2  = (const float*)d_in[18];
    const float* bf2  = (const float*)d_in[19];
    float* outp = (float*)d_out;

    char* ws = (char*)d_ws;
    size_t off = 0;
    auto carve = [&](size_t bytes) -> void* {
        void* p = ws + off;
        off = (off + bytes + 255) & ~(size_t)255;
        return p;
    };
    bf16_t* wqT  = (bf16_t*)carve(192 * 192 * 2);
    bf16_t* wkvT = (bf16_t*)carve(384 * 192 * 2);
    bf16_t* wpT  = (bf16_t*)carve(192 * 192 * 2);
    bf16_t* wf1T = (bf16_t*)carve(768 * 192 * 2);
    bf16_t* wf2T = (bf16_t*)carve(192 * 768 * 2);
    size_t qkv_bytes = (size_t)BWIN * NHEAD * NPAD * HD * 2;
    bf16_t* qg   = (bf16_t*)carve(qkv_bytes);
    bf16_t* kg   = (bf16_t*)carve(qkv_bytes);
    bf16_t* vtg  = (bf16_t*)carve(qkv_bytes);
    bf16_t* owin = (bf16_t*)carve((size_t)BWIN * NPAD * CDIM * 2);
    float*  xbuf = (float*)carve((size_t)ROWS * CDIM * 4);
    bf16_t* xn2  = (bf16_t*)carve((size_t)ROWS * CDIM * 2);

    /* weight transpose + bf16 convert */
    k_wt<<<(192 * 192 + 255) / 256, 256, 0, stream>>>(wq,  wqT,  192, 192);
    k_wt<<<(192 * 384 + 255) / 256, 256, 0, stream>>>(wkv, wkvT, 192, 384);
    k_wt<<<(192 * 192 + 255) / 256, 256, 0, stream>>>(wp,  wpT,  192, 192);
    k_wt<<<(192 * 768 + 255) / 256, 256, 0, stream>>>(wf1, wf1T, 192, 768);
    k_wt<<<(768 * 192 + 255) / 256, 256, 0, stream>>>(wf2, wf2T, 768, 192);

    /* LN + shift + window partition + QKV projection */
    k_qkv<<<BWIN, 256, 2 * NPAD * CDIM * 2, stream>>>(
        x1, x2, g1, b1, g2, b2, wqT, bq, wkvT, bkv, qg, kg, vtg);

    /* attention per (window, head) */
    size_t attn_lds = (size_t)NPAD * HD * 2 * 2 + 4 * 16 * NPAD * 4 + 4 * 16 * NPAD * 2;
    k_attn<<<BWIN * NHEAD, 128, attn_lds, stream>>>(qg, kg, vtg, bias_table, mask, owin);

    /* projection + reverse shift + residual */
    k_proj<<<BWIN, 256, NPAD * CDIM * 2, stream>>>(owin, wpT, bp, x1, xbuf);

    /* LN2 */
    k_ln2<<<ROWS / 8, 256, 0, stream>>>(xbuf, gn2, bn2, xn2);

    /* fused MLP + final residual */
    size_t mlp_lds = (size_t)64 * CDIM * 2 + (size_t)64 * HID * 2;
    k_mlp<<<ROWS / 64, 256, mlp_lds, stream>>>(xn2, wf1T, bf1, wf2T, bf2, xbuf, outp);
}
